// SpikingConv2D_43593918054971
// MI455X (gfx1250) — compile-verified, matrix operand order checked by reading, not run on hardware
//
#include <hip/hip_runtime.h>

// ---------------------------------------------------------------------------
// SpikingConv2D as implicit GEMM on CDNA5 WMMA (bf16 in, f32 accumulate)
//   M = B*H*W = 100352 = 784 * 128,  N = F = 128,  K = 9*C = 1152 = 36*32
// K-order permuted to (kh, kw, c): every 32-wide K slice has fixed (kh,kw),
// so A-staging becomes contiguous 32-channel NHWC bf16 copies (async-to-LDS).
// LDS rows padded to 80 B (40 bf16) for conflict-free ds_load_b128 fragments.
// ---------------------------------------------------------------------------

typedef __attribute__((ext_vector_type(16))) __bf16        v16bf;
typedef __attribute__((ext_vector_type(8)))  float         v8f;
typedef __attribute__((ext_vector_type(4)))  int           v4i;
typedef __attribute__((ext_vector_type(4)))  unsigned int  u32x4;
typedef __attribute__((ext_vector_type(8)))  unsigned int  u32x8;

#define B_    32
#define C_    128
#define H_    56
#define W_    56
#define F_    128
#define HW_   (H_ * W_)          // 3136
#define MTOT  (B_ * HW_)         // 100352
#define KTOT  (C_ * 9)           // 1152
#define KSTEP 32
#define NKS   (KTOT / KSTEP)     // 36
#define LDA   40                 // padded LDS row length in bf16 (80 B)

#if __has_builtin(__builtin_amdgcn_global_load_async_to_lds_b128) && \
    __has_builtin(__builtin_amdgcn_s_wait_asynccnt)
#define HAVE_ASYNC 1
#else
#define HAVE_ASYNC 0
#endif

// round-to-nearest-even f32 -> bf16
static __device__ __forceinline__ unsigned short f2bf(float f) {
    unsigned int u = __builtin_bit_cast(unsigned int, f);
    u += 0x7FFFu + ((u >> 16) & 1u);
    return (unsigned short)(u >> 16);
}

// 16-byte global -> LDS move (ASYNCcnt path when available)
static __device__ __forceinline__ void ld16(const unsigned short* g,
                                            unsigned short* l) {
#if HAVE_ASYNC
    __builtin_amdgcn_global_load_async_to_lds_b128(
        (__attribute__((address_space(1))) v4i*)(g),
        (__attribute__((address_space(3))) v4i*)(l), 0, 0);
#else
    *(u32x4*)l = *(const u32x4*)g;
#endif
}

static __device__ __forceinline__ void wait_async0() {
#if HAVE_ASYNC
    __builtin_amdgcn_s_wait_asynccnt(0);
#endif
}

// Load one 16x32 bf16 fragment from LDS (row stride LDA bf16), matching the
// CDNA5 16-bit A layout (lanes 0-15: K0-7,K16-23; lanes 16-31: K8-15,K24-31).
// B is staged [N][K] so the identical loader produces the B fragment.
static __device__ __forceinline__ v16bf load_frag(const unsigned short* smem,
                                                  int row, int laneHi) {
    const char* base = (const char*)smem + row * (LDA * 2) + laneHi * 16;
    u32x4 c0 = *(const u32x4*)(base);
    u32x4 c1 = *(const u32x4*)(base + 32);
    u32x8 t  = __builtin_shufflevector(c0, c1, 0, 1, 2, 3, 4, 5, 6, 7);
    return __builtin_bit_cast(v16bf, t);
}

// ---------------------------------------------------------------------------
// Prep 1: W [k=(c,kh,kw)][F] f32 -> Wt [N=128][k'=(kh,kw,c)=1152] bf16
// ---------------------------------------------------------------------------
__global__ __launch_bounds__(256) void wt_transpose_bf16(
    const float* __restrict__ W, unsigned short* __restrict__ Wt) {
    int idx = blockIdx.x * 256 + threadIdx.x;   // 0 .. 147455
    int n  = idx / KTOT;
    int kp = idx - n * KTOT;        // permuted k' = r*128 + c
    int r  = kp >> 7;               // kh*3+kw
    int c  = kp & 127;
    Wt[n * KTOT + kp] = f2bf(W[(c * 9 + r) * F_ + n]);
}

// ---------------------------------------------------------------------------
// Prep 2: tj [B,C,H,W] f32 -> tjT [B,H,W,C] bf16 (LDS-tiled 32x32 transpose)
// tj is read exactly once -> non-temporal loads.
// ---------------------------------------------------------------------------
__global__ __launch_bounds__(256) void tj_transpose_bf16(
    const float* __restrict__ tj, unsigned short* __restrict__ tjT) {
    __shared__ float tile[32][33];
    const int p0  = blockIdx.x * 32;          // global pixel base (same image)
    const int c0  = blockIdx.y * 32;
    const int b   = p0 / HW_;
    const int hw0 = p0 - b * HW_;
    const int j = threadIdx.x & 31;
    const int i = threadIdx.x >> 5;           // 0..7
    const float* src = tj + (size_t)b * C_ * HW_ + hw0;
#pragma unroll
    for (int ii = 0; ii < 4; ++ii)
        tile[i + ii * 8][j] =
            __builtin_nontemporal_load(&src[(size_t)(c0 + i + ii * 8) * HW_ + j]);
    __syncthreads();
    unsigned short* dst = tjT + (size_t)p0 * C_ + c0;
#pragma unroll
    for (int ii = 0; ii < 4; ++ii) {
        int pp = i + ii * 8;
        dst[(size_t)pp * C_ + j] = f2bf(tile[j][pp]);
    }
}

// ---------------------------------------------------------------------------
// Main: one block = 128(M) x 128(N) tile, 8 waves (4 M x 2 N), each wave
// 32x64 via 2x4 v_wmma_f32_16x16x32_bf16; double-buffered async LDS staging.
// ---------------------------------------------------------------------------
__global__ __launch_bounds__(256) void spiking_conv_wmma(
    const unsigned short* __restrict__ tjT, // [B,H,W,C] bf16
    const unsigned short* __restrict__ Wt,  // [N=128][K'=1152] bf16
    const float* __restrict__ Di,           // [9,128] f32 (row 0 used)
    float* __restrict__ out) {              // [B,H,W,F] f32 == [M,128]
    __shared__ unsigned short As[2 * 128 * LDA];  // 2 x 10 KB
    __shared__ unsigned short Bs[2 * 128 * LDA];  // 2 x 10 KB

    const int tid = threadIdx.x;
    const int m0  = blockIdx.x * 128;

    // staging mapping: each thread moves 32 B of A and 32 B of B per step
    const int smi   = tid & 127;    // tile row (M for A, N for B)
    const int skseg = tid >> 7;     // 16-element half of the 32-wide K slice

    const int m   = m0 + smi;
    const int bb  = m / HW_;
    const int rem = m - bb * HW_;
    const int h   = rem / W_;
    const int w   = rem - h * W_;

    // compute mapping
    const int lane   = tid & 31;
    const int waveId = tid >> 5;
    const int wm     = (waveId & 3) * 32;
    const int wn     = (waveId >> 2) * 64;
    const int laneLo = lane & 15;
    const int laneHi = lane >> 4;

    auto stage = [&](int ks, int buf) {
        const int boff = buf ? 128 * LDA : 0;
        // ---- A: 32 channels at shifted pixel, fixed (kh,kw) per step ------
        const int r  = ks >> 2;               // kh*3 + kw
        const int kh = r / 3;
        const int kw = r - kh * 3;
        const int c0 = ((ks & 3) << 5) + (skseg << 4);
        const int ih = h + kh - 1;
        const int iw = w + kw - 1;
        unsigned short* dstA = &As[boff + smi * LDA + (skseg << 4)];
        if ((unsigned)ih < (unsigned)H_ && (unsigned)iw < (unsigned)W_) {
            const unsigned short* srcA =
                tjT + ((size_t)bb * HW_ + ih * W_ + iw) * C_ + c0;
            ld16(srcA, dstA);
            ld16(srcA + 8, dstA + 8);
        } else {
            *(u32x4*)dstA       = (u32x4)0u;
            *(u32x4*)(dstA + 8) = (u32x4)0u;
        }
        // ---- B: contiguous slice of Wt ------------------------------------
        const unsigned short* srcB = Wt + smi * KTOT + ks * KSTEP + (skseg << 4);
        unsigned short* dstB = &Bs[boff + smi * LDA + (skseg << 4)];
        ld16(srcB, dstB);
        ld16(srcB + 8, dstB + 8);
    };

    v8f acc[2][4] = {};

    stage(0, 0);
    for (int ks = 0; ks < NKS; ++ks) {
        wait_async0();          // my buffer-(ks&1) loads are in LDS
        __syncthreads();        // everyone's are
        if (ks + 1 < NKS) stage(ks + 1, (ks + 1) & 1);

        const unsigned short* Ab = &As[(ks & 1) ? 128 * LDA : 0];
        const unsigned short* Bb = &Bs[(ks & 1) ? 128 * LDA : 0];
        v16bf a[2], b[4];
#pragma unroll
        for (int mi = 0; mi < 2; ++mi)
            a[mi] = load_frag(Ab, wm + mi * 16 + laneLo, laneHi);
#pragma unroll
        for (int ni = 0; ni < 4; ++ni)
            b[ni] = load_frag(Bb, wn + ni * 16 + laneLo, laneHi);

#pragma unroll
        for (int mi = 0; mi < 2; ++mi)
#pragma unroll
            for (int ni = 0; ni < 4; ++ni)
                acc[mi][ni] = __builtin_amdgcn_wmma_f32_16x16x32_bf16(
                    false, a[mi], false, b[ni], (short)0, acc[mi][ni],
                    false, false);
    }

    // epilogue: ti = min(y + (1 - D_i[0][n]), 1.0), NHWC non-temporal store
#pragma unroll
    for (int ni = 0; ni < 4; ++ni) {
        const int n   = wn + ni * 16 + laneLo;
        const float th = 1.0f - Di[n];
#pragma unroll
        for (int mi = 0; mi < 2; ++mi) {
            v8f c = acc[mi][ni];
#pragma unroll
            for (int i = 0; i < 8; ++i) {
                int mm = wm + mi * 16 + laneHi * 8 + i;
                __builtin_nontemporal_store(fminf(c[i] + th, 1.0f),
                                            &out[(size_t)(m0 + mm) * F_ + n]);
            }
        }
    }
}

// ---------------------------------------------------------------------------
extern "C" void kernel_launch(void* const* d_in, const int* in_sizes, int n_in,
                              void* d_out, int out_size, void* d_ws,
                              size_t ws_size, hipStream_t stream) {
    (void)in_sizes; (void)n_in; (void)out_size; (void)ws_size;
    const float* tj = (const float*)d_in[0];   // [32,128,56,56]
    const float* W  = (const float*)d_in[1];   // [1152,128]
    const float* Di = (const float*)d_in[2];   // [9,128]
    float* out = (float*)d_out;                // [32,56,56,128] == [M,128]

    unsigned short* Wt  = (unsigned short*)d_ws;            // 294,912 B
    unsigned short* tjT = Wt + (size_t)F_ * KTOT;           // 25,690,112 B

    wt_transpose_bf16<<<(KTOT * F_) / 256, 256, 0, stream>>>(W, Wt);
    tj_transpose_bf16<<<dim3(MTOT / 32, C_ / 32), 256, 0, stream>>>(tj, tjT);
    spiking_conv_wmma<<<MTOT / 128, 256, 0, stream>>>(tjT, Wt, Di, out);
}
